// Histcounts_21311627723520
// MI455X (gfx1250) — compile-verified, matrix operand order checked by reading, not run on hardware
//
#include <hip/hip_runtime.h>

// ---------------------------------------------------------------------------
// Per-row histogram (tf.histogram_fixed_width semantics), MI455X / gfx1250.
//
//   B=4096 rows, N=16384 values/row, NBINS=100, range [0,256).
//   idx = clip(floor(100 * clip(trunc(x),0,256) / 256), 0, 99)
//       = min((25 * min(max((int)x,0),256)) >> 6, 99)      (bit-exact)
//
// Memory-bound: 256 MiB streamed once -> ~11.5 us floor at 23.3 TB/s.
// Design: 1 block (8 wave32) per row; NT b128 streaming loads + prefetch;
// 32 LDS histogram copies (per wave x lane&3) updated with ds_add_u32;
// copy-reduction done with V_WMMA_F32_16X16X32_F16 (K=32 == #copies,
// f16 staging exact since per-copy counts <= 512 < 2048).
// ---------------------------------------------------------------------------

typedef __attribute__((ext_vector_type(4)))  float    v4f;
typedef __attribute__((ext_vector_type(16))) _Float16 v16h;
typedef __attribute__((ext_vector_type(8)))  float    v8f;

#define NBINS    100
#define NBPAD    112          // 7 groups of 16 bins (WMMA N-tiles)
#define NGROUP   (NBPAD / 16) // 7
#define NCOPY    32           // histogram copies == WMMA K dimension
#define ROW_N    16384
#define THREADS  256          // 8 wave32
#define HI_EDGE  256

__global__ __launch_bounds__(THREADS)
void hist_rows_kernel(const float* __restrict__ x, float* __restrict__ out) {
  __shared__ unsigned int hist[NCOPY * NBPAD];   // 14336 B

  const int row  = blockIdx.x;
  const int tid  = threadIdx.x;
  const int wave = tid >> 5;
  const int lane = tid & 31;

  // --- zero the privatized histograms -------------------------------------
  #pragma unroll
  for (int i = tid; i < NCOPY * NBPAD; i += THREADS) hist[i] = 0u;
  __syncthreads();

  // --- streaming accumulation ---------------------------------------------
  // copy id: (wave, lane&3) -> 32 copies; <=8 lanes ever share one copy.
  const int sub = ((wave << 2) | (lane & 3)) * NBPAD;
  const v4f* __restrict__ rp = (const v4f*)(x + (size_t)row * ROW_N);
  const int NIT = ROW_N / (THREADS * 4);         // 16 iterations, 4 KB/block/iter

  #pragma unroll 4
  for (int it = 0; it < NIT; ++it) {
    const int vi = it * THREADS + tid;
    if (vi + 2 * THREADS < ROW_N / 4)            // stay inside the row
      __builtin_prefetch((const void*)(rp + vi + 2 * THREADS), 0, 0);
    const v4f v = __builtin_nontemporal_load(rp + vi);  // global_load_b128 NT
    #pragma unroll
    for (int e = 0; e < 4; ++e) {
      int xi = (int)v[e];                        // trunc toward zero (x >= 0)
      xi = xi < 0 ? 0 : (xi > HI_EDGE ? HI_EDGE : xi);
      unsigned idx = ((unsigned)(25 * xi)) >> 6; // floor(100*xi/256), exact
      if (idx > (NBINS - 1)) idx = NBINS - 1;
      atomicAdd(&hist[sub + idx], 1u);           // ds_add_u32
    }
  }
  __syncthreads();

  // --- WMMA reduction over the 32 copies ----------------------------------
  // D(16x16) = ones(16x32,f16) x B(32x16,f16) + 0 ; B[k][n] = hist[k][g*16+n].
  // 16-bit B layout (wave32): lane holds column n = lane&15; VGPR half j maps
  // to K = j       (lanes 0-15, j<8),  K = j+8  (lanes 0-15, j>=8)
  //        j+8     (lanes16-31, j<8),  K = j+16 (lanes16-31, j>=8)
  // All 8 waves compute identically (EXEC all-1s for WMMA); wave 0 stores.
  v16h a;
  #pragma unroll
  for (int j = 0; j < 16; ++j) a[j] = (_Float16)1.0f;

  const int n = lane & 15;
  #pragma unroll
  for (int g = 0; g < NGROUP; ++g) {
    const int bin = g * 16 + n;
    v16h b;
    #pragma unroll
    for (int j = 0; j < 16; ++j) {
      const int k = (lane < 16) ? (j < 8 ? j : j + 8)
                                : (j < 8 ? j + 8 : j + 16);
      b[j] = (_Float16)(float)hist[k * NBPAD + bin];  // <=512, exact in f16
    }
    v8f c = {};
    // 8 args: (neg_a, A, neg_b, B, c_mod, C, reuse_a, reuse_b)
    c = __builtin_amdgcn_wmma_f32_16x16x32_f16(
        false, a, false, b, (short)0, c, false, false);
    // D row M=0 lives in VGPR0 of lanes 0..15 (N = lane).
    if (tid < 16 && bin < NBINS)
      out[(size_t)row * NBINS + bin] = c[0];
  }
}

extern "C" void kernel_launch(void* const* d_in, const int* in_sizes, int n_in,
                              void* d_out, int out_size, void* d_ws, size_t ws_size,
                              hipStream_t stream) {
  (void)n_in; (void)out_size; (void)d_ws; (void)ws_size;
  const float* x  = (const float*)d_in[0];
  float*      out = (float*)d_out;
  const int rows  = in_sizes[0] / ROW_N;   // 4096
  hist_rows_kernel<<<rows, THREADS, 0, stream>>>(x, out);
}
